// LatexDecoder_25434796327414
// MI455X (gfx1250) — matrix operand order repaired and checked by reading, not compiled, not run on hardware
//
#include <hip/hip_runtime.h>
#include <hip/hip_bf16.h>
#include <stddef.h>
#include <stdint.h>

// ---------------------------------------------------------------------------
// Persistent-kernel attention decoder for MI455X (gfx1250, wave32, WMMA).
//
// 16 workgroups x 256 threads (8 wave32 per WG), co-resident (one per WGP).
// All 128 decode steps run inside one kernel; phases separated by a
// device-scope atomic grid barrier. Step-invariant weight slices (~125 KB/WG)
// are staged ONCE into LDS (320 KB/WGP) — via the Tensor Data Mover where the
// builtin exists — so every per-step B-fragment comes from ds_load instead of
// re-streaming L2. All GEMMs run on v_wmma_f32_16x16x32_f16, f32 accumulate.
// ---------------------------------------------------------------------------

typedef __attribute__((ext_vector_type(16))) _Float16 v16h;
typedef __attribute__((ext_vector_type(8)))  _Float16 v8h;
typedef __attribute__((ext_vector_type(8)))  float    v8f;
typedef __attribute__((ext_vector_type(4)))  unsigned v4u;
typedef __attribute__((ext_vector_type(4)))  int      v4i_t;
typedef __attribute__((ext_vector_type(8)))  int      v8i_t;

#define NWG 16
#define NT  256
#define B_  16
#define L_  256
#define D_  256
#define H_  256
#define E_  256
#define V_  172
#define T_  128

// Padded LDS row strides (halves): +8 halves (16 B) so consecutive rows are
// offset by 4 banks -> conflict-free b128 reads across 16 lanes.
#define P512 520
#define P256 264

// LDS layout (bytes)
#define LDS_RED   0                         // float red[8][16][16]    (8192)
#define LDS_SA    8192                      // float s_a[256]
#define LDS_SB    9216                      // float s_b[256]
#define LDS_SRED  10240                     // float s_red[256]
#define LDS_WIH   11264                     // 64 rows x P512 halves   (66560)
#define LDS_WHH   (LDS_WIH + 64 * P512 * 2) // 64 rows x P256 halves   (33792)
#define LDS_W2    (LDS_WHH + 64 * P256 * 2) // 16 rows x P256 halves   (8448)
#define LDS_W3    (LDS_W2  + 16 * P256 * 2) // 16 rows x P512 halves   (16640)
#define LDS_TOTAL (LDS_W3  + 16 * P512 * 2) // = 136704 bytes

#if defined(__has_builtin)
#  if __has_builtin(__builtin_amdgcn_tensor_load_to_lds) && \
      __has_builtin(__builtin_amdgcn_s_wait_tensorcnt)
#    define USE_TDM 1
#  endif
#endif
#ifndef USE_TDM
#  define USE_TDM 0
#endif

// ---- WMMA fragment helpers (layouts per CDNA5 ISA 7.12.2, wave32) ----------

// A-matrix 16x32 f16, row-major global source [16 x lda]:
// lane<16: VGPR0..3 = K(k0..k0+7), VGPR4..7 = K(k0+16..k0+23) of row M=lane;
// lane>=16 shifted by +8/+24.
__device__ __forceinline__ v16h load_a16x32(const _Float16* A, int lda, int k0, int lane) {
  int m  = lane & 15;
  int kh = lane >> 4;
  const _Float16* p = A + (size_t)m * lda + k0 + kh * 8;
  v8h lo = *(const v8h*)(p);
  v8h hi = *(const v8h*)(p + 16);
  v16h r;
#pragma unroll
  for (int i = 0; i < 8; ++i) { r[i] = lo[i]; r[i + 8] = hi[i]; }
  return r;
}

// B-matrix 32x16 f16 for out = X @ W^T, W slice staged in LDS row-major with
// padded stride ldp. Lane L holds column (L&15), K = k0+(L>>4)*8 .. +8 and
// k0+16+(L>>4)*8 .. +8 (2x 16B ds loads, 16B aligned).
__device__ __forceinline__ v16h load_b_lds(const _Float16* Wb, int ldp, int k0, int lane) {
  const _Float16* p = Wb + (lane & 15) * ldp + k0 + (lane >> 4) * 8;
  v8h lo = *(const v8h*)(p);
  v8h hi = *(const v8h*)(p + 16);
  v16h r;
#pragma unroll
  for (int i = 0; i < 8; ++i) { r[i] = lo[i]; r[i + 8] = hi[i]; }
  return r;
}

__device__ __forceinline__ v8f wmma32(v16h a, v16h b, v8f c) {
  return __builtin_amdgcn_wmma_f32_16x16x32_f16(false, a, false, b, (short)0, c, false, false);
}

__device__ __forceinline__ float sigm(float x) { return 1.f / (1.f + __expf(-x)); }

#if USE_TDM
// ---- Tensor Data Mover 2-D tile load with LDS row padding ------------------
// D# per CDNA5 ISA 8.3/8.4: group0 {count=1, lds_addr, global_addr, type=2},
// group1 {data_size=2B, pad_enable, pad_interval, pad_amount, dims/strides}.
__device__ __forceinline__ void tdm_load_2d(const void* gsrc, unsigned lds_byte,
                                            unsigned width, unsigned rows,
                                            unsigned stride,
                                            unsigned pad_int, unsigned pad_amt) {
  unsigned long long ga = (unsigned long long)(uintptr_t)gsrc;
  v4u g0;
  g0.x = 1u;                                                   // count=1 (user D#)
  g0.y = lds_byte;                                             // lds_addr
  g0.z = (unsigned)ga;                                         // global_addr[31:0]
  g0.w = (unsigned)((ga >> 32) & 0x01FFFFFFull) | (2u << 30);  // [56:32] | type=2
  v8i_t g1;
  g1[0] = (int)((1u << 16) | (1u << 20) |                      // data_size=2B, pad_en
                (pad_int << 22) | (pad_amt << 25));
  g1[1] = (int)((width & 0xFFFFu) << 16);                      // tensor_dim0[15:0]
  g1[2] = (int)(((width >> 16) & 0xFFFFu) |                    // tensor_dim0[31:16]
                ((rows & 0xFFFFu) << 16));                     // tensor_dim1[15:0]
  g1[3] = (int)(((rows >> 16) & 0xFFFFu) |                     // tensor_dim1[31:16]
                ((width & 0xFFFFu) << 16));                    // tile_dim0
  g1[4] = (int)(rows & 0xFFFFu);                               // tile_dim1 (tile_dim2=0)
  g1[5] = (int)stride;                                         // tensor_dim0_stride lo
  g1[6] = 0;
  g1[7] = 0;
  v4i_t z4 = (v4i_t)0;
#if __clang_major__ >= 23
  v8i_t z8 = (v8i_t)0;
  __builtin_amdgcn_tensor_load_to_lds(g0, g1, z4, z4, z8, 0);
#else
  __builtin_amdgcn_tensor_load_to_lds(g0, g1, z4, z4, 0);
#endif
}
#endif

// ---- device-scope grid barrier (bar[0]=count, bar[1]=generation) -----------
__device__ __forceinline__ void grid_barrier(unsigned* bar) {
  __syncthreads();
  if (threadIdx.x == 0) {
    __threadfence();
    unsigned* cnt = bar;
    unsigned* gen = bar + 1;
    unsigned g = __hip_atomic_load(gen, __ATOMIC_RELAXED, __HIP_MEMORY_SCOPE_AGENT);
    unsigned prev = __hip_atomic_fetch_add(cnt, 1u, __ATOMIC_ACQ_REL, __HIP_MEMORY_SCOPE_AGENT);
    if (prev == NWG - 1) {
      __hip_atomic_store(cnt, 0u, __ATOMIC_RELAXED, __HIP_MEMORY_SCOPE_AGENT);
      __hip_atomic_fetch_add(gen, 1u, __ATOMIC_RELEASE, __HIP_MEMORY_SCOPE_AGENT);
    } else {
      while (__hip_atomic_load(gen, __ATOMIC_ACQUIRE, __HIP_MEMORY_SCOPE_AGENT) == g)
        __builtin_amdgcn_s_sleep(2);
    }
    __threadfence();
  }
  __syncthreads();
}

// ---- workgroup reductions (blockDim.x == 256) ------------------------------
__device__ __forceinline__ float blk_max(float v, float* s) {
  int t = threadIdx.x;
  s[t] = v; __syncthreads();
#pragma unroll
  for (int o = 128; o > 0; o >>= 1) {
    if (t < o) s[t] = fmaxf(s[t], s[t + o]);
    __syncthreads();
  }
  float r = s[0]; __syncthreads();
  return r;
}
__device__ __forceinline__ float blk_sum(float v, float* s) {
  int t = threadIdx.x;
  s[t] = v; __syncthreads();
#pragma unroll
  for (int o = 128; o > 0; o >>= 1) {
    if (t < o) s[t] += s[t + o];
    __syncthreads();
  }
  float r = s[0]; __syncthreads();
  return r;
}

// ---- parameter block -------------------------------------------------------
struct Params {
  const float* imgs;      // [B,L,D] f32
  const int*   tgt;       // [B,T]
  const float* b_ih;      // [4H]
  const float* b_hh;      // [4H]
  const float* beta;      // [D]
  const _Float16* W_ih_h; // [4H, H+E]
  const _Float16* W_hh_h; // [4H, H]
  const _Float16* W2_h;   // [D, H]
  const _Float16* W3_h;   // [H, H+D]
  const _Float16* Wout_h; // [V, H]
  const _Float16* emb_h;  // [V, E]
  const float* encproj;   // [B,L,D] f32 (step-invariant W1 projection)
  _Float16* h0;           // [B,H] ping
  _Float16* h1;           // [B,H] pong
  float*    c;            // [B,H] cell state (WG-column-exclusive)
  _Float16* inp;          // [B, E+H] = [emb_t || o_{t-1}]
  _Float16* cat2;         // [B, H+D] = [h_t || ctx_t]
  float*    q;            // [B,D] = h_t @ W2^T
  _Float16* o16;          // [B,H]
  unsigned* bar;          // grid barrier state
  float*    out;          // [B,T,V]
};

// ---- one-time prep: f32->f16 weight conversion, barrier reset --------------
__global__ __launch_bounds__(256) void k_prep(
    const float* Wih, const float* Whh, const float* W2, const float* W3,
    const float* Wout, const float* emb,
    _Float16* Wih_h, _Float16* Whh_h, _Float16* W2_h, _Float16* W3_h,
    _Float16* Wout_h, _Float16* emb_h, unsigned* bar) {
  int idx = blockIdx.x * 256 + threadIdx.x;
  if (idx == 0) bar[0] = 0u;
  if (idx < 4 * H_ * (H_ + E_)) Wih_h[idx]  = (_Float16)Wih[idx];
  if (idx < 4 * H_ * H_)        Whh_h[idx]  = (_Float16)Whh[idx];
  if (idx < D_ * H_)            W2_h[idx]   = (_Float16)W2[idx];
  if (idx < H_ * (H_ + D_))     W3_h[idx]   = (_Float16)W3[idx];
  if (idx < V_ * H_)            Wout_h[idx] = (_Float16)Wout[idx];
  if (idx < V_ * E_)            emb_h[idx]  = (_Float16)emb[idx];
}

// ---- mean-pooled encoder output --------------------------------------------
__global__ __launch_bounds__(256) void k_mean(const float* imgs, float* mean) {
  int b = blockIdx.x, d = threadIdx.x;
  float s = 0.f;
#pragma unroll 4
  for (int l = 0; l < L_; ++l) s += imgs[(size_t)(b * L_ + l) * D_ + d];
  mean[b * D_ + d] = s * (1.f / L_);
}

// ---- decoder init state + step-0 embedding ---------------------------------
__global__ __launch_bounds__(256) void k_init(
    const float* mean, const float* whW, const float* whb, const float* wcW,
    const float* wcb, const float* woW, const float* wob, const float* emb,
    const int* tgt, _Float16* h0, float* c, _Float16* inp) {
  __shared__ float sm[256];
  int b = blockIdx.x, d = threadIdx.x;
  sm[d] = mean[b * D_ + d];
  __syncthreads();
  float ah = 0.f, ac = 0.f, ao = 0.f;
#pragma unroll 4
  for (int k = 0; k < D_; ++k) {
    float m = sm[k];
    ah += m * whW[d * D_ + k];
    ac += m * wcW[d * D_ + k];
    ao += m * woW[d * D_ + k];
  }
  h0[b * H_ + d] = (_Float16)tanhf(ah + whb[d]);
  c[b * H_ + d]  = tanhf(ac + wcb[d]);
  inp[b * 512 + 256 + d] = (_Float16)tanhf(ao + wob[d]);
  int v = tgt[b * T_ + 0];
  inp[b * 512 + d] = (_Float16)emb[v * E_ + d];
}

// ---- enc_proj[b,l,e] = sum_d imgs[b,l,d] * W1[e,d] (one-time) --------------
__global__ __launch_bounds__(256) void k_encproj(const float* imgs, const float* W1, float* ep) {
  __shared__ float row[256];
  int bl = blockIdx.x;       // b*L + l
  int e = threadIdx.x;
  row[e] = imgs[(size_t)bl * D_ + e];
  __syncthreads();
  float a = 0.f;
#pragma unroll 4
  for (int d = 0; d < D_; ++d) a += row[d] * W1[e * D_ + d];
  ep[(size_t)bl * D_ + e] = a;
}

// ---- persistent decode loop ------------------------------------------------
__global__ __launch_bounds__(256) void k_main(Params P) {
  const int tid  = threadIdx.x;
  const int wg   = blockIdx.x;         // 0..15
  const int lane = tid & 31;
  const int wave = tid >> 5;           // 0..7

  extern __shared__ char smem[];
  float (*red)[16][16] = (float (*)[16][16])(smem + LDS_RED);
  float* s_a   = (float*)(smem + LDS_SA);
  float* s_b   = (float*)(smem + LDS_SB);
  float* s_red = (float*)(smem + LDS_SRED);
  _Float16* ldsWih = (_Float16*)(smem + LDS_WIH);  // 4 blocks of 16 rows x P512
  _Float16* ldsWhh = (_Float16*)(smem + LDS_WHH);  // 4 blocks of 16 rows x P256
  _Float16* ldsW2  = (_Float16*)(smem + LDS_W2);   // 16 rows x P256
  _Float16* ldsW3  = (_Float16*)(smem + LDS_W3);   // 16 rows x P512

  // ---- stage this WG's step-invariant weight slices into LDS (once) --------
#if USE_TDM
  if (wave == 0) {
#pragma unroll
    for (int j = 0; j < 4; ++j) {
      tdm_load_2d(P.W_ih_h + ((size_t)j * H_ + wg * 16) * 512,
                  (unsigned)(uintptr_t)(ldsWih + j * 16 * P512),
                  512, 16, 512, /*pad_int=*/7, /*pad_amt=*/3);
      tdm_load_2d(P.W_hh_h + ((size_t)j * H_ + wg * 16) * 256,
                  (unsigned)(uintptr_t)(ldsWhh + j * 16 * P256),
                  256, 16, 256, /*pad_int=*/6, /*pad_amt=*/3);
    }
    tdm_load_2d(P.W2_h + (size_t)wg * 16 * 256,
                (unsigned)(uintptr_t)ldsW2, 256, 16, 256, 6, 3);
    tdm_load_2d(P.W3_h + (size_t)wg * 16 * 512,
                (unsigned)(uintptr_t)ldsW3, 512, 16, 512, 7, 3);
    __builtin_amdgcn_s_wait_tensorcnt(0);
  }
#else
  for (int i = tid; i < 64 * 512; i += NT) {
    int r = i >> 9, col = i & 511;
    ldsWih[r * P512 + col] =
        P.W_ih_h[((size_t)((r >> 4) * H_) + wg * 16 + (r & 15)) * 512 + col];
  }
  for (int i = tid; i < 64 * 256; i += NT) {
    int r = i >> 8, col = i & 255;
    ldsWhh[r * P256 + col] =
        P.W_hh_h[((size_t)((r >> 4) * H_) + wg * 16 + (r & 15)) * 256 + col];
  }
  for (int i = tid; i < 16 * 256; i += NT) {
    int r = i >> 8, col = i & 255;
    ldsW2[r * P256 + col] = P.W2_h[(size_t)(wg * 16 + r) * 256 + col];
  }
  for (int i = tid; i < 16 * 512; i += NT) {
    int r = i >> 9, col = i & 511;
    ldsW3[r * P512 + col] = P.W3_h[(size_t)(wg * 16 + r) * 512 + col];
  }
#endif
  __syncthreads();

  for (int t = 0; t < T_; ++t) {
    const _Float16* hprev = (t & 1) ? P.h1 : P.h0;
    _Float16*       hcur  = (t & 1) ? P.h0 : P.h1;

    // ---- Phase 1: gates = [emb||o] @ W_ih^T + h @ W_hh^T; LSTM update -----
    // WG owns H-columns [wg*16, wg*16+16). Wave (j = w&3) computes gate j's
    // 16x16 tile; waves j and j+4 split the K=768 reduction. B from LDS.
    {
      int j = wave & 3, kh = wave >> 2;
      const _Float16* Wih_b = ldsWih + j * 16 * P512;
      const _Float16* Whh_b = ldsWhh + j * 16 * P256;
      v8f acc = {};
      if (kh == 0) {
#pragma unroll
        for (int kk = 0; kk < 12; ++kk) {        // K 0..383 (inp)
          v16h a = load_a16x32(P.inp, 512, kk * 32, lane);
          v16h b = load_b_lds(Wih_b, P512, kk * 32, lane);
          acc = wmma32(a, b, acc);
        }
      } else {
#pragma unroll
        for (int kk = 0; kk < 4; ++kk) {         // K 384..511 (inp)
          v16h a = load_a16x32(P.inp, 512, 384 + kk * 32, lane);
          v16h b = load_b_lds(Wih_b, P512, 384 + kk * 32, lane);
          acc = wmma32(a, b, acc);
        }
#pragma unroll
        for (int kk = 0; kk < 8; ++kk) {         // K 512..767 (h_prev)
          v16h a = load_a16x32(hprev, 256, kk * 32, lane);
          v16h b = load_b_lds(Whh_b, P256, kk * 32, lane);
          acc = wmma32(a, b, acc);
        }
      }
      {
        int n = lane & 15, mh = (lane >> 4) * 8;
#pragma unroll
        for (int r = 0; r < 8; ++r) red[wave][mh + r][n] = acc[r];
      }
      __syncthreads();
      // elementwise LSTM on this WG's exclusive column slice
      int m = tid >> 4, col = tid & 15;
      int nc = wg * 16 + col;
      float gi = red[0][m][col] + red[4][m][col] + P.b_ih[0 * H_ + nc] + P.b_hh[0 * H_ + nc];
      float gf = red[1][m][col] + red[5][m][col] + P.b_ih[1 * H_ + nc] + P.b_hh[1 * H_ + nc];
      float gg = red[2][m][col] + red[6][m][col] + P.b_ih[2 * H_ + nc] + P.b_hh[2 * H_ + nc];
      float go = red[3][m][col] + red[7][m][col] + P.b_ih[3 * H_ + nc] + P.b_hh[3 * H_ + nc];
      int ci = m * H_ + nc;
      float cn = sigm(gf) * P.c[ci] + sigm(gi) * tanhf(gg);
      float hn = sigm(go) * tanhf(cn);
      P.c[ci] = cn;
      _Float16 hh = (_Float16)hn;
      hcur[ci] = hh;
      P.cat2[m * 512 + nc] = hh;     // [h || ctx] buffer, h part
      __syncthreads();
    }
    grid_barrier(P.bar);

    // ---- Phase 2: q = h_new @ W2^T (one tile per WG, K split over waves) --
    {
      v8f acc = {};
      v16h a = load_a16x32(hcur, 256, wave * 32, lane);
      v16h b = load_b_lds(ldsW2, P256, wave * 32, lane);
      acc = wmma32(a, b, acc);
      int n = lane & 15, mh = (lane >> 4) * 8;
#pragma unroll
      for (int r = 0; r < 8; ++r) red[wave][mh + r][n] = acc[r];
      __syncthreads();
      int m = tid >> 4, col = tid & 15;
      float s = 0.f;
#pragma unroll
      for (int w = 0; w < 8; ++w) s += red[w][m][col];
      P.q[m * H_ + wg * 16 + col] = s;
      __syncthreads();
    }
    grid_barrier(P.bar);

    // ---- Phase 3: score/softmax/ctx, batch b = wg (VALU, tanh-heavy) ------
    {
      int b = wg;
      s_a[tid] = P.q[b * D_ + tid];
      s_b[tid] = P.beta[tid];
      __syncthreads();
      const float* ep = P.encproj + (size_t)(b * L_ + tid) * D_;  // row l=tid
      float sc = 0.f;
#pragma unroll 4
      for (int d = 0; d < D_; ++d)
        sc += s_b[d] * tanhf(ep[d] + s_a[d]);
      float mx = blk_max(sc, s_red);
      float e  = __expf(sc - mx);
      float sm = blk_sum(e, s_red);
      s_a[tid] = e;                     // unnormalized attention
      __syncthreads();
      float inv = 1.f / sm;
      const float* im = P.imgs + (size_t)b * L_ * D_ + tid;       // column d=tid
      float cx = 0.f;
#pragma unroll 4
      for (int l = 0; l < L_; ++l) cx += s_a[l] * im[(size_t)l * D_];
      P.cat2[b * 512 + 256 + tid] = (_Float16)(cx * inv);
      __syncthreads();
    }
    grid_barrier(P.bar);

    // ---- Phase 4: o = tanh(cat2 @ W3^T); prefetch next embedding ----------
    {
      v8f acc = {};
#pragma unroll
      for (int kk = 0; kk < 2; ++kk) {
        int k0 = wave * 64 + kk * 32;
        v16h a = load_a16x32(P.cat2, 512, k0, lane);
        v16h b = load_b_lds(ldsW3, P512, k0, lane);
        acc = wmma32(a, b, acc);
      }
      int n = lane & 15, mh = (lane >> 4) * 8;
#pragma unroll
      for (int r = 0; r < 8; ++r) red[wave][mh + r][n] = acc[r];
      __syncthreads();
      int m = tid >> 4, col = tid & 15;
      float s = 0.f;
#pragma unroll
      for (int w = 0; w < 8; ++w) s += red[w][m][col];
      _Float16 oh = (_Float16)tanhf(s);
      P.o16[m * H_ + wg * 16 + col] = oh;
      P.inp[m * 512 + 256 + wg * 16 + col] = oh;   // o part of next step input
      if (t + 1 < T_) {                            // emb part of next input
        int bb = tid >> 4, e = wg * 16 + (tid & 15);
        int vtok = P.tgt[bb * T_ + t + 1];
        P.inp[bb * 512 + e] = P.emb_h[vtok * E_ + e];
      }
      __syncthreads();
    }
    grid_barrier(P.bar);

    // ---- Phase 5: logits softmax, batch b = wg ----------------------------
    {
      int b = wg, v = tid;
      float sc = -1e30f;
      if (v < V_) {
        const _Float16* orow = P.o16 + b * H_;
        const _Float16* wrow = P.Wout_h + v * H_;
        float acc = 0.f;
#pragma unroll 8
        for (int k = 0; k < H_; ++k) acc += (float)orow[k] * (float)wrow[k];
        sc = acc;
      }
      float mx = blk_max(sc, s_red);
      float e  = (v < V_) ? __expf(sc - mx) : 0.f;
      float sm = blk_sum(e, s_red);
      if (v < V_) P.out[((size_t)b * T_ + t) * V_ + v] = e / sm;
    }
    // no barrier needed: next Phase 1 only reads buffers synced at barrier 4
  }
}

// ---------------------------------------------------------------------------
extern "C" void kernel_launch(void* const* d_in, const int* in_sizes, int n_in,
                              void* d_out, int out_size, void* d_ws, size_t ws_size,
                              hipStream_t stream) {
  (void)in_sizes; (void)n_in; (void)out_size; (void)ws_size;
  const float* imgs = (const float*)d_in[0];
  const int*   tgt  = (const int*)d_in[1];
  const float* emb  = (const float*)d_in[2];
  const float* W_ih = (const float*)d_in[3];
  const float* W_hh = (const float*)d_in[4];
  const float* b_ih = (const float*)d_in[5];
  const float* b_hh = (const float*)d_in[6];
  const float* W1   = (const float*)d_in[7];
  const float* W2   = (const float*)d_in[8];
  const float* W3   = (const float*)d_in[9];
  const float* Wout = (const float*)d_in[10];
  const float* beta = (const float*)d_in[11];
  const float* wh_W = (const float*)d_in[12];
  const float* wh_b = (const float*)d_in[13];
  const float* wc_W = (const float*)d_in[14];
  const float* wc_b = (const float*)d_in[15];
  const float* wo_W = (const float*)d_in[16];
  const float* wo_b = (const float*)d_in[17];
  float* out = (float*)d_out;

  // carve workspace (all offsets 1 KiB aligned; total ~6.3 MB)
  char* ws = (char*)d_ws;
  size_t off = 0;
  auto alloc = [&](size_t bytes) -> char* {
    char* p = ws + off;
    off = (off + bytes + 1023) & ~(size_t)1023;
    return p;
  };
  unsigned*  bar     = (unsigned*)alloc(64);
  _Float16*  W_ih_h  = (_Float16*)alloc((size_t)4 * H_ * (H_ + E_) * 2);
  _Float16*  W_hh_h  = (_Float16*)alloc((size_t)4 * H_ * H_ * 2);
  _Float16*  W2_h    = (_Float16*)alloc((size_t)D_ * H_ * 2);
  _Float16*  W3_h    = (_Float16*)alloc((size_t)H_ * (H_ + D_) * 2);
  _Float16*  Wout_h  = (_Float16*)alloc((size_t)V_ * H_ * 2);
  _Float16*  emb_h   = (_Float16*)alloc((size_t)V_ * E_ * 2);
  float*     encproj = (float*)alloc((size_t)B_ * L_ * D_ * 4);
  float*     mean    = (float*)alloc((size_t)B_ * D_ * 4);
  _Float16*  h0      = (_Float16*)alloc((size_t)B_ * H_ * 2);
  _Float16*  h1      = (_Float16*)alloc((size_t)B_ * H_ * 2);
  float*     cbuf    = (float*)alloc((size_t)B_ * H_ * 4);
  _Float16*  inp     = (_Float16*)alloc((size_t)B_ * 512 * 2);
  _Float16*  cat2    = (_Float16*)alloc((size_t)B_ * 512 * 2);
  float*     q       = (float*)alloc((size_t)B_ * D_ * 4);
  _Float16*  o16     = (_Float16*)alloc((size_t)B_ * H_ * 2);

  // allow >64KB dynamic LDS for the persistent kernel (capture-safe: no
  // stream work is enqueued by attribute setting)
  hipFuncSetAttribute((const void*)k_main,
                      hipFuncAttributeMaxDynamicSharedMemorySize, LDS_TOTAL);

  // one-time prep (re-run every call; ws is not preserved across poisoning)
  k_prep<<<(4 * H_ * (H_ + E_) + 255) / 256, 256, 0, stream>>>(
      W_ih, W_hh, W2, W3, Wout, emb,
      W_ih_h, W_hh_h, W2_h, W3_h, Wout_h, emb_h, bar);
  k_mean<<<B_, 256, 0, stream>>>(imgs, mean);
  k_init<<<B_, 256, 0, stream>>>(mean, wh_W, wh_b, wc_W, wc_b, wo_W, wo_b,
                                 emb, tgt, h0, cbuf, inp);
  k_encproj<<<B_ * L_, 256, 0, stream>>>(imgs, W1, encproj);

  Params P;
  P.imgs = imgs; P.tgt = tgt; P.b_ih = b_ih; P.b_hh = b_hh; P.beta = beta;
  P.W_ih_h = W_ih_h; P.W_hh_h = W_hh_h; P.W2_h = W2_h; P.W3_h = W3_h;
  P.Wout_h = Wout_h; P.emb_h = emb_h; P.encproj = encproj;
  P.h0 = h0; P.h1 = h1; P.c = cbuf; P.inp = inp; P.cat2 = cat2;
  P.q = q; P.o16 = o16; P.bar = bar; P.out = out;

  k_main<<<NWG, NT, LDS_TOTAL, stream>>>(P);
}
